// CPCPredictor_5729486373529
// MI455X (gfx1250) — matrix operand order repaired and checked by reading, not compiled, not run on hardware
//
#include <hip/hip_runtime.h>
#include <hip/hip_bf16.h>

// ---------------------------------------------------------------------------
// CPC loss for MI455X (gfx1250, wave32, WMMA + async global->LDS copies).
//  Stage 0: fp32 -> bf16 conversion of pred_c and W (one pass).
//  Stage 1: per-k GEMM  Wc = pred_c @ W[k]^T + b[k]
//           v_wmma_f32_16x16x32_bf16, 128x128 block tile, 32x64 wave tile,
//           double-buffered LDS filled with GLOBAL_LOAD_ASYNC_TO_LDS_B128.
//  Stage 2: per-(b,l) 11 logits (1 pos + 10 hashed negatives), log-softmax,
//           deterministic block partial sums.
//  Stage 3: single-wave finalize -> out[0]=loss, out[1..12]=acc[k].
// ---------------------------------------------------------------------------

typedef __attribute__((ext_vector_type(16))) __bf16 v16bf;
typedef __attribute__((ext_vector_type(8)))  float  v8f;

union ABFrag { v16bf v; uint4 q[2]; };

#define CPC_B   144
#define CPC_T   140
#define CPC_D   256
#define CPC_C   256
#define CPC_K   12
#define CPC_L   128   // T - K
#define CPC_S   12
#define CPC_U   12
#define CPC_NN  10    // num_negatives
#define CPC_M   (CPC_B * CPC_L)          // 18432 GEMM rows
#define ST3_BLOCKS (CPC_M / 8)           // 2304 blocks, 8 waves each

__device__ __forceinline__ unsigned hash_u32(unsigned x) {
    x ^= x >> 16; x *= 0x7feb352dU;
    x ^= x >> 15; x *= 0x846ca68bU;
    x ^= x >> 16;
    return x;
}

// 16-byte async global->LDS copy (ASYNCcnt-tracked, bypasses VGPR staging).
__device__ __forceinline__ void async_copy16(const void* gptr, void* lptr) {
    unsigned           lds = (unsigned)(unsigned long long)(uintptr_t)lptr; // low 32b = LDS offset
    unsigned long long ga  = (unsigned long long)(uintptr_t)gptr;
    asm volatile("global_load_async_to_lds_b128 %0, %1, off"
                 :: "v"(lds), "v"(ga) : "memory");
}

__device__ __forceinline__ void wait_async0() {
#if __has_builtin(__builtin_amdgcn_s_wait_asynccnt)
    __builtin_amdgcn_s_wait_asynccnt(0);
#else
    asm volatile("s_wait_asynccnt 0" ::: "memory");
#endif
}

// ---------------- Stage 0: conversions -------------------------------------
__global__ void cvt_c_kernel(const float* __restrict__ c, __bf16* __restrict__ dst) {
    int idx = blockIdx.x * 256 + threadIdx.x;            // < B*L*C
    int b   = idx / (CPC_L * CPC_C);
    int rem = idx - b * (CPC_L * CPC_C);
    dst[idx] = (__bf16)c[(size_t)b * CPC_T * CPC_C + rem];
}

__global__ void cvt_w_kernel(const float* __restrict__ w, __bf16* __restrict__ dst) {
    int idx = blockIdx.x * 256 + threadIdx.x;            // < K*D*C
    dst[idx] = (__bf16)w[idx];
}

// ---------------- Stage 1: bf16 WMMA GEMM ----------------------------------
// A: (M=18432, C=256) bf16 row-major (pred_c)
// Bm: (D=256, C=256) bf16 row-major (W[k]); output col d uses row d of Bm.
// out: (M, D) fp32,  out = A @ Bm^T + bias
__global__ __launch_bounds__(256) void gemm_bf16_wmma(
        const __bf16* __restrict__ A, const __bf16* __restrict__ Bm,
        const float* __restrict__ bias, float* __restrict__ out) {
    // double-buffered tiles, 40-elem pitch (80 B) keeps 16B alignment, no conflicts
    __shared__ __align__(16) __bf16 As[2][128][40];   // 10240 B per buffer
    __shared__ __align__(16) __bf16 Bs[2][128][40];   // 10240 B per buffer

    const int t     = threadIdx.x;
    const int lane  = t & 31;
    const int wv    = t >> 5;
    const int waveM = wv >> 1;        // 0..3  -> 32 rows each
    const int waveN = wv & 1;         // 0..1  -> 64 cols each
    const int mBase = blockIdx.y * 128;
    const int nBase = blockIdx.x * 128;

    v8f acc[2][4] = {};

    // issue one 128x32 A tile + 128x32 B tile as async copies (4 x b128/thread)
    auto issue_tile = [&](int buf, int kb) {
#pragma unroll
        for (int i = 0; i < 2; ++i) {
            int idx = t + i * 256;            // 0..511
            int row = idx >> 2, c8 = idx & 3;
            async_copy16(&A [(size_t)(mBase + row) * CPC_C + kb + c8 * 8], &As[buf][row][c8 * 8]);
            async_copy16(&Bm[(size_t)(nBase + row) * CPC_C + kb + c8 * 8], &Bs[buf][row][c8 * 8]);
        }
    };

    issue_tile(0, 0);
    wait_async0();
    __syncthreads();

    int buf = 0;
#pragma unroll 1
    for (int kb = 0; kb < CPC_C; kb += 32) {
        if (kb + 32 < CPC_C) issue_tile(buf ^ 1, kb + 32);   // prefetch next K-tile

        // Fragment assembly per ISA 7.12.2 bf16 layouts.
        ABFrag afr[2], bfr[4];
        const int K0a = (lane & 16) ? 8 : 0;    // A: lanes16-31 hold K 8-15 / 24-31
        const int K0b = (lane & 16) ? 8 : 0;    // B: lanes16-31 hold K 8-15 / 24-31
#pragma unroll
        for (int mt = 0; mt < 2; ++mt) {
            int row = waveM * 32 + mt * 16 + (lane & 15);
            afr[mt].q[0] = *(const uint4*)&As[buf][row][K0a];
            afr[mt].q[1] = *(const uint4*)&As[buf][row][K0a + 16];
        }
#pragma unroll
        for (int nt = 0; nt < 4; ++nt) {
            int row = waveN * 64 + nt * 16 + (lane & 15);
            bfr[nt].q[0] = *(const uint4*)&Bs[buf][row][K0b];
            bfr[nt].q[1] = *(const uint4*)&Bs[buf][row][K0b + 16];
        }
#pragma unroll
        for (int mt = 0; mt < 2; ++mt)
#pragma unroll
            for (int nt = 0; nt < 4; ++nt)
                acc[mt][nt] = __builtin_amdgcn_wmma_f32_16x16x32_bf16(
                    false, afr[mt].v, false, bfr[nt].v,
                    (short)0, acc[mt][nt], false, false);

        wait_async0();      // prefetched tile landed
        __syncthreads();    // everyone done reading buf / writing buf^1
        buf ^= 1;
    }

    // C/D layout: lane N = lane&15 ; VGPR r -> M = r (+8 for lanes 16-31)
#pragma unroll
    for (int mt = 0; mt < 2; ++mt)
#pragma unroll
        for (int nt = 0; nt < 4; ++nt) {
            int col = nBase + waveN * 64 + nt * 16 + (lane & 15);
            float bv = bias[col];
            int mrow0 = mBase + waveM * 32 + mt * 16 + ((lane & 16) ? 8 : 0);
#pragma unroll
            for (int r = 0; r < 8; ++r)
                out[(size_t)(mrow0 + r) * CPC_D + col] = acc[mt][nt][r] + bv;
        }
}

// ---------------- Stage 2: logits + log-softmax ----------------------------
__global__ __launch_bounds__(256) void cpc_logits_kernel(
        const float* __restrict__ z, const float* __restrict__ wc, int k,
        float* __restrict__ lossp, float* __restrict__ hitp) {
    const int lane = threadIdx.x & 31;
    const int wv   = threadIdx.x >> 5;
    const int g    = blockIdx.x * 8 + wv;     // (b,l) pair, 0..18431
    const int b    = g >> 7;                  // L = 128
    const int l    = g & 127;
    const int s    = b / CPC_U;
    const int u    = b % CPC_U;

    const float* wrow = wc + (size_t)g * CPC_D + lane * 8;
    const float4 w0 = *(const float4*)wrow;
    const float4 w1 = *(const float4*)(wrow + 4);

    float f[CPC_NN + 1];
#pragma unroll 1
    for (int n = 0; n <= CPC_NN; ++n) {
        int bb, ll;
        if (n == 0) { bb = b; ll = l; }
        else {
            // deterministic stand-in for jax.random (no runtime check available)
            unsigned hb = hash_u32(((unsigned)(k * CPC_U + u) * 16u + (unsigned)n) * 0x9E3779B9u);
            unsigned hs = hash_u32(((((unsigned)((k * CPC_S + s) * CPC_U + u)) * 11u
                                     + (unsigned)n) * 128u + (unsigned)l) ^ 0x85EBCA6Bu);
            bb = s * CPC_U + (int)(hb % (unsigned)CPC_U);
            ll = (int)((hs % (unsigned)CPC_L) + (unsigned)l) & (CPC_L - 1);
        }
        const float* zr = z + ((size_t)bb * CPC_T + (k + 1 + ll)) * CPC_D + lane * 8;
        const float4 z0 = *(const float4*)zr;
        const float4 z1 = *(const float4*)(zr + 4);
        float p = z0.x * w0.x + z0.y * w0.y + z0.z * w0.z + z0.w * w0.w
                + z1.x * w1.x + z1.y * w1.y + z1.z * w1.z + z1.w * w1.w;
#pragma unroll
        for (int m = 16; m >= 1; m >>= 1) p += __shfl_xor(p, m);
        f[n] = p * 0.0625f;                   // 1/sqrt(256)
    }

    float fmax = f[0];
    int hit = 1;
#pragma unroll
    for (int n = 1; n <= CPC_NN; ++n) {
        if (f[n] > f[0]) hit = 0;
        if (f[n] > fmax) fmax = f[n];
    }
    float sum = 0.f;
#pragma unroll
    for (int n = 0; n <= CPC_NN; ++n) sum += __expf(f[n] - fmax);
    float logp0 = f[0] - fmax - __logf(sum);

    __shared__ float sl[8], sh[8];
    if (lane == 0) { sl[wv] = -logp0; sh[wv] = (float)hit; }
    __syncthreads();
    if (threadIdx.x == 0) {
        float a = 0.f, h = 0.f;
#pragma unroll
        for (int i = 0; i < 8; ++i) { a += sl[i]; h += sh[i]; }
        lossp[blockIdx.x] = a;
        hitp[blockIdx.x]  = h;
    }
}

// ---------------- Stage 3: finalize -----------------------------------------
__global__ void finalize_kernel(const float* __restrict__ lossp,
                                const float* __restrict__ hitp,
                                float* __restrict__ out) {
    const int lane = threadIdx.x;             // 32 threads
    float ls = 0.f;
    if (lane < CPC_K) {
        const float* lp = lossp + lane * ST3_BLOCKS;
        const float* hp = hitp  + lane * ST3_BLOCKS;
        float hs = 0.f;
        for (int i = 0; i < ST3_BLOCKS; ++i) { ls += lp[i]; hs += hp[i]; }
        out[1 + lane] = hs / (float)(CPC_S * CPC_U * CPC_L);
    }
#pragma unroll
    for (int m = 16; m >= 1; m >>= 1) ls += __shfl_xor(ls, m);
    if (lane == 0) out[0] = ls / (float)(CPC_K * CPC_S * CPC_U * CPC_L);
}

// ---------------------------------------------------------------------------
extern "C" void kernel_launch(void* const* d_in, const int* in_sizes, int n_in,
                              void* d_out, int out_size, void* d_ws, size_t ws_size,
                              hipStream_t stream) {
    (void)in_sizes; (void)n_in; (void)out_size; (void)ws_size;
    const float* z    = (const float*)d_in[0];   // (B,T,D)
    const float* c    = (const float*)d_in[1];   // (B,T,C)
    const float* W    = (const float*)d_in[2];   // (K,D,C)
    const float* bias = (const float*)d_in[3];   // (K,D)
    // d_in[4] = num_negatives (==10, hardcoded)

    char* ws = (char*)d_ws;
    size_t off = 0;
    __bf16* cbf = (__bf16*)(ws + off); off += (size_t)CPC_M * CPC_C * sizeof(__bf16);         // 9.4 MB
    __bf16* wbf = (__bf16*)(ws + off); off += (size_t)CPC_K * CPC_D * CPC_C * sizeof(__bf16); // 1.6 MB
    float*  wc  = (float*)(ws + off);  off += (size_t)CPC_M * CPC_D * sizeof(float);          // 18.9 MB
    float*  lossp = (float*)(ws + off); off += (size_t)CPC_K * ST3_BLOCKS * sizeof(float);
    float*  hitp  = (float*)(ws + off); off += (size_t)CPC_K * ST3_BLOCKS * sizeof(float);

    cvt_c_kernel<<<(CPC_M * CPC_C) / 256, 256, 0, stream>>>(c, cbf);
    cvt_w_kernel<<<(CPC_K * CPC_D * CPC_C) / 256, 256, 0, stream>>>(W, wbf);

    for (int k = 0; k < CPC_K; ++k) {
        gemm_bf16_wmma<<<dim3(CPC_D / 128, CPC_M / 128), 256, 0, stream>>>(
            cbf, wbf + (size_t)k * CPC_D * CPC_C, bias + (size_t)k * CPC_D, wc);
        cpc_logits_kernel<<<ST3_BLOCKS, 256, 0, stream>>>(
            z, wc, k, lossp + (size_t)k * ST3_BLOCKS, hitp + (size_t)k * ST3_BLOCKS);
    }
    finalize_kernel<<<1, 32, 0, stream>>>(lossp, hitp, (float*)d_out);
}
